// MoeExpertTokenRemap_15822659519278
// MI455X (gfx1250) — compile-verified, matrix operand order checked by reading, not run on hardware
//
#include <hip/hip_runtime.h>
#include <hip/hip_bf16.h>

// MoE expert->device token remap for MI455X (gfx1250, wave32).
//   T=32768 tokens, E=256 experts, K=8 selected, D=8 devices, r=2 pair-reduce.
// Outputs (concatenated in d_out, float):
//   remapped [D, T, E/D]  = 8*32768*32 floats
//   reduced_mask [T/2, D] = 16384*8 floats (0.0/1.0)

#define T_TOK 32768
#define E_EXP 256
#define K_SEL 8
#define D_DEV 8

typedef __attribute__((ext_vector_type(8))) int v8i;
typedef __attribute__((ext_vector_type(4))) float v4f;

// ws layout (ints): perm[256] | ident[1] | pad | bwords[32*32] @ int offset 272
#define WS_BW_OFF 272

// ---------------------------------------------------------------------------
// Kernel 1: tiny setup — dev_of_expert = argmax(mapping), stable argsort perm,
// identity detection, and precomputed WMMA B-matrix words (mapping one-hot in
// IU8 B layout, per lane: 4 K-blocks x 8 packed byte-words).
// ---------------------------------------------------------------------------
__global__ void moe_setup_perm(const int* __restrict__ mapping,
                               int* __restrict__ ws) {
  __shared__ int dev_s[E_EXP];
  __shared__ int ok_s;
  int* perm_ws = ws;
  int* ident_flag = ws + E_EXP;
  int* bws = ws + WS_BW_OFF;

  const int e = threadIdx.x;
  int d = 0, best = mapping[e * D_DEV];
#pragma unroll
  for (int j = 1; j < D_DEV; ++j) {
    int v = mapping[e * D_DEV + j];
    if (v > best) { best = v; d = j; }
  }
  dev_s[e] = d;
  if (e == 0) ok_s = 1;
  __syncthreads();
  // stable rank: (#experts with smaller device) + (#earlier experts, same device)
  int pos = 0;
  for (int ep = 0; ep < E_EXP; ++ep) {
    int de = dev_s[ep];
    pos += (de < d || (de == d && ep < e)) ? 1 : 0;
  }
  perm_ws[pos] = e;                 // perm[rank] = expert  (jnp.argsort stable)
  if (pos != e) atomicAnd(&ok_s, 0);
  __syncthreads();
  if (e == 0) ident_flag[0] = ok_s;

  // B-matrix words for the IU8 WMMA mask kernel. Lane l holds B[k, n=l&15]
  // bytes for K set selected by half = l>>4 (ISA 7.12.2 byte packing).
  if (e < 32) {
    const int l = e;
    const int half = l >> 4;
    const int n = l & 15;
    for (int blk = 0; blk < 4; ++blk) {
      for (int g = 0; g < 4; ++g) {
        for (int vj = 0; vj < 2; ++vj) {
          unsigned w = 0;
          for (int b = 0; b < 4; ++b) {
            const int k = 16 * g + 8 * half + 4 * vj + b;
            const int ex = blk * 64 + k;
            unsigned byte = 0;
            if (n < D_DEV) byte = ((unsigned)mapping[ex * D_DEV + n]) & 0xFFu;
            w |= byte << (8 * b);
          }
          bws[l * 32 + blk * 8 + 2 * g + vj] = (int)w;
        }
      }
    }
  }
}

// ---------------------------------------------------------------------------
// Kernel 2: main streaming remap. One wave per token, one lane per 8 experts.
// Fast path (perm == identity): 2x b128 NT load, mask in registers, 2x b128
// NT store. ~64 MB total traffic -> HBM-bound (~2.8 us at 23.3 TB/s).
// ---------------------------------------------------------------------------
__global__ void moe_remap(const float* __restrict__ topk,
                          const int* __restrict__ meta,
                          const int* __restrict__ ws,
                          float* __restrict__ out) {
  __shared__ int perm_s[E_EXP];
  const int tid = threadIdx.x;
  perm_s[tid] = ws[tid];
  __syncthreads();

  const int lane = tid & 31;
  const int wave = tid >> 5;
  const int t = blockIdx.x * 8 + wave;

  const int4* m4 = (const int4*)(meta + (size_t)t * K_SEL);
  const int4 ma = m4[0], mb = m4[1];
  const int mv[8] = {ma.x, ma.y, ma.z, ma.w, mb.x, mb.y, mb.z, mb.w};

  float v[8];
  if (ws[E_EXP]) {  // identity permutation fast path (uniform branch)
    // coalesced: lane covers experts [lane*8, lane*8+8)
    const v4f* src = (const v4f*)(topk + (size_t)t * E_EXP + lane * 8);
    if (t + 64 < T_TOK)  // stream-ahead prefetch (gfx1250 global_prefetch_b8)
      __builtin_prefetch(topk + (size_t)(t + 64) * E_EXP + lane * 8, 0, 1);
    const v4f r0 = __builtin_nontemporal_load(src);      // single-use stream
    const v4f r1 = __builtin_nontemporal_load(src + 1);
    v[0] = r0.x; v[1] = r0.y; v[2] = r0.z; v[3] = r0.w;
    v[4] = r1.x; v[5] = r1.y; v[6] = r1.z; v[7] = r1.w;
    unsigned sel = 0;
#pragma unroll
    for (int k = 0; k < 8; ++k)
      if ((mv[k] >> 3) == lane) sel |= 1u << (mv[k] & 7);
#pragma unroll
    for (int i = 0; i < 8; ++i)
      if (!((sel >> i) & 1u)) v[i] = 0.0f;
  } else {
    // general permutation: gather through perm
#pragma unroll
    for (int i = 0; i < 8; ++i) {
      const int e = perm_s[lane * 8 + i];
      bool hit = false;
#pragma unroll
      for (int k = 0; k < 8; ++k) hit = hit || (mv[k] == e);
      v[i] = hit ? topk[(size_t)t * E_EXP + e] : 0.0f;
    }
  }

  // output column group lane*8..lane*8+7 -> device d = lane/4, slot (lane%4)*8
  const int d = lane >> 2;
  const int joff = (lane & 3) * 8;
  float* dst = out + ((size_t)d * T_TOK + t) * (E_EXP / D_DEV) + joff;
  const v4f s0 = {v[0], v[1], v[2], v[3]};
  const v4f s1 = {v[4], v[5], v[6], v[7]};
  __builtin_nontemporal_store(s0, (v4f*)dst);        // write-once stream
  __builtin_nontemporal_store(s1, (v4f*)(dst + 4));
}

// ---------------------------------------------------------------------------
// Kernel 3: dispatch mask via V_WMMA_I32_16X16X64_IU8.
//   count[t,d] = sum_e sel[t,e] * mapping[e,d]   (u8 x u8 -> i32)
// A built from an 8-word selection bitmap, expanded nibble->bytes with the
// (nib * 0x00204081) & 0x01010101 spread. B preloaded from ws (b128 loads).
// C layout: VGPR g -> M=g (lanes 0-15) / M=g+8 (lanes 16-31), N = lane&15.
// ---------------------------------------------------------------------------
__global__ void moe_mask_wmma(const int* __restrict__ meta,
                              const int* __restrict__ ws,
                              float* __restrict__ mask_out) {
  const int tid = threadIdx.x;
  const int lane = tid & 31;
  const int half = lane >> 4;   // A/B K byte-group selector
  const int n = lane & 15;      // B/C column (device); A row for meta loads
  const int waveG = blockIdx.x * (blockDim.x >> 5) + (tid >> 5);
  const int wavesTotal = gridDim.x * (blockDim.x >> 5);

  // Load precomputed B: 32 words per lane = 8x int4.
  const int4* bp = (const int4*)(ws + WS_BW_OFF + lane * 32);
  v8i Bm[4];
#pragma unroll
  for (int blk = 0; blk < 4; ++blk) {
    const int4 b0 = bp[blk * 2];
    const int4 b1 = bp[blk * 2 + 1];
    Bm[blk] = (v8i){b0.x, b0.y, b0.z, b0.w, b1.x, b1.y, b1.z, b1.w};
  }

  const int groups = T_TOK >> 4;  // 16 tokens per WMMA group
  for (int grp = waveG; grp < groups; grp += wavesTotal) {
    const int T0 = grp << 4;
    const int tok = T0 + n;       // this lane's A row (M = n)
    const int4* m4 = (const int4*)(meta + (size_t)tok * K_SEL);
    const int4 ma = m4[0], mb = m4[1];
    const int mv[8] = {ma.x, ma.y, ma.z, ma.w, mb.x, mb.y, mb.z, mb.w};

    // 256-bit selection bitmap over experts (8 u32, registers only)
    unsigned bm[8];
#pragma unroll
    for (int wi = 0; wi < 8; ++wi) bm[wi] = 0u;
#pragma unroll
    for (int k2 = 0; k2 < 8; ++k2) {
      const int m = mv[k2];
      const unsigned bit = 1u << (m & 31);
      const int w = m >> 5;
#pragma unroll
      for (int wi = 0; wi < 8; ++wi)
        if (w == wi) bm[wi] |= bit;
    }

    v8i C = {};
#pragma unroll
    for (int blk = 0; blk < 4; ++blk) {
      // expand bitmap nibbles into A's byte layout
      v8i A;
#pragma unroll
      for (int g = 0; g < 4; ++g) {
#pragma unroll
        for (int vj = 0; vj < 2; ++vj) {
          const int word = blk * 2 + (g >> 1);             // static index
          const int shift = ((g & 1) * 16) + 8 * half + 4 * vj;
          const unsigned nib = (bm[word] >> shift) & 0xFu;
          A[2 * g + vj] = (int)((nib * 0x00204081u) & 0x01010101u);
        }
      }
      // (sgn_a, A, sgn_b, B, C, reuse_a, reuse_b) -> v_wmma_i32_16x16x64_iu8
      C = __builtin_amdgcn_wmma_i32_16x16x64_iu8(false, A, false, Bm[blk], C,
                                                 false, false);
    }

    // pair-reduce consecutive tokens, threshold, write [T/2, D]
    if (n < D_DEV) {
#pragma unroll
      for (int p = 0; p < 4; ++p) {
        const int cnt = C[2 * p] + C[2 * p + 1];            // tokens M=2p,2p+1
        const int row = (T0 >> 1) + p + (half ? 4 : 0);     // half 1 -> M+8
        mask_out[(size_t)row * D_DEV + n] = (cnt > 0) ? 1.0f : 0.0f;
      }
    }
  }
}

// ---------------------------------------------------------------------------
extern "C" void kernel_launch(void* const* d_in, const int* in_sizes, int n_in,
                              void* d_out, int out_size, void* d_ws, size_t ws_size,
                              hipStream_t stream) {
  const float* topk   = (const float*)d_in[0];  // [1,1,T,E] f32
  const int* mapping  = (const int*)d_in[1];    // [1,1,E,D] i32 one-hot
  const int* meta     = (const int*)d_in[2];    // [1,1,T,K] i32
  float* out = (float*)d_out;
  float* mask_out = out + (size_t)D_DEV * T_TOK * (E_EXP / D_DEV);
  int* ws = (int*)d_ws;

  moe_setup_perm<<<1, E_EXP, 0, stream>>>(mapping, ws);
  moe_remap<<<T_TOK / 8, 256, 0, stream>>>(topk, meta, ws, out);
  moe_mask_wmma<<<128, 256, 0, stream>>>(meta, ws, mask_out);
}
